// GATNoBidirectionalGraphModel_74105365725625
// MI455X (gfx1250) — matrix operand order repaired and checked by reading, not compiled, not run on hardware
//
#include <hip/hip_runtime.h>

#define N_NODES 100000
#define N_EDGES 1600000
#define E_TOT   (N_EDGES + N_NODES)
#define F_IN    128
#define H1_     128
#define H2_     64
#define FC1_    128
#define C_      10
#define M_      20000
#define SLOPE_  0.2f

typedef __attribute__((ext_vector_type(16))) __bf16 v16bf;
typedef __attribute__((ext_vector_type(8)))  __bf16 v8bf;
typedef __attribute__((ext_vector_type(8)))  float  v8f;

static __device__ __forceinline__ unsigned short bfbits(float f) {
    unsigned u = __builtin_bit_cast(unsigned, f);
    return (unsigned short)((u + 0x7FFFu + ((u >> 16) & 1u)) >> 16);   // RNE
}
static __device__ __forceinline__ __bf16 f2bf(float f) {
    return __builtin_bit_cast(__bf16, bfbits(f));
}

// order-preserving float <-> uint key (for atomicMax on floats)
static __device__ __forceinline__ unsigned fkey(float f) {
    unsigned u = __builtin_bit_cast(unsigned, f);
    return (u & 0x80000000u) ? ~u : (u | 0x80000000u);
}
static __device__ __forceinline__ float funkey(unsigned k) {
    unsigned u = (k & 0x80000000u) ? (k ^ 0x80000000u) : ~k;
    return __builtin_bit_cast(float, u);
}

// ---- f32 -> bf16 bulk convert (4 elems / thread, packed 64-bit stores) ----------------
__global__ void cvt_bf16(const float* __restrict__ in, __bf16* __restrict__ out, long long total)
{
    long long i = 4ll * (blockIdx.x * (long long)blockDim.x + threadIdx.x);
    if (i >= total) return;
    float4 v = *(const float4*)(in + i);
    unsigned lo = (unsigned)bfbits(v.x) | ((unsigned)bfbits(v.y) << 16);
    unsigned hi = (unsigned)bfbits(v.z) | ((unsigned)bfbits(v.w) << 16);
    uint2 p; p.x = lo; p.y = hi;
    *(uint2*)(out + i) = p;
}

// ---- weight convert + transpose: BT[n*K+k] = bf16(B[k*Nc+n]) --------------------------
__global__ void cvt_bf16_T(const float* __restrict__ B, __bf16* __restrict__ BT, int K, int Nc)
{
    int i = blockIdx.x * blockDim.x + threadIdx.x;
    if (i >= K * Nc) return;
    int k = i / Nc, n = i - k * Nc;
    BT[(size_t)n * K + k] = f2bf(B[i]);
}

// -------- WMMA GEMM: C[M x NT*16] = A[MxK](bf16) @ BT[NT*16 x K]^T (+bias, relu) -------
// One wave computes a full 16 x (NT*16) row strip; A fragment loaded once per K-block.
template <int K, int NT>
__global__ void gemm_bf16_wmma(const __bf16* __restrict__ A, const __bf16* __restrict__ BT,
                               float* __restrict__ C, const float* __restrict__ bias,
                               int M, int relu)
{
    const int wave  = threadIdx.x >> 5;
    const int lane  = threadIdx.x & 31;
    const int mtile = blockIdx.x * (blockDim.x >> 5) + wave;
    if (mtile * 16 >= M) return;                 // wave-uniform; EXEC all-ones inside

    const int mloc = lane & 15;
    const int ksel = lane >> 4;
    const int Nc   = NT * 16;

    v8f acc[NT];
    #pragma unroll
    for (int nt = 0; nt < NT; ++nt) acc[nt] = (v8f){};

    #pragma unroll
    for (int kb = 0; kb < K; kb += 32) {
        // A 16x32 bf16 fragment: V0-3 -> K = kb+8*ksel+[0..7]; V4-7 -> +16
        const __bf16* Ap = A + (size_t)(mtile * 16 + mloc) * K + kb + ksel * 8;
        v8bf a0 = *(const v8bf*)(Ap);
        v8bf a1 = *(const v8bf*)(Ap + 16);
        v16bf a = __builtin_shufflevector(a0, a1, 0,1,2,3,4,5,6,7,8,9,10,11,12,13,14,15);
        #pragma unroll
        for (int nt = 0; nt < NT; ++nt) {
            // B 32x16 fragment: V0-3 -> K = kb+8*ksel+[0..7]; V4-7 -> +16
            const __bf16* Bp = BT + (size_t)(nt * 16 + mloc) * K + kb + ksel * 8;
            v8bf b0 = *(const v8bf*)(Bp);
            v8bf b1 = *(const v8bf*)(Bp + 16);
            v16bf b = __builtin_shufflevector(b0, b1, 0,1,2,3,4,5,6,7,8,9,10,11,12,13,14,15);
            acc[nt] = __builtin_amdgcn_wmma_f32_16x16x32_bf16(
                          false, a, false, b, (short)0, acc[nt], false, false);
        }
    }

    const int mbase = ksel * 8;                  // C/D: lanes 16-31 hold rows M=8..15
    #pragma unroll
    for (int nt = 0; nt < NT; ++nt) {
        const int col = nt * 16 + mloc;
        const float bv = bias ? bias[col] : 0.0f;
        #pragma unroll
        for (int v = 0; v < 8; ++v) {
            float r = acc[nt][v] + bv;
            if (relu) r = fmaxf(r, 0.0f);
            C[(size_t)(mtile * 16 + mbase + v) * Nc + col] = r;
        }
    }
}

// -------- per-node attention logits: alpha_s = h.a_src, alpha_d = h.a_dst (wave/node) --
__global__ void alpha_kernel(const float* __restrict__ h,
                             const float* __restrict__ asrc, const float* __restrict__ adst,
                             float* __restrict__ als, float* __restrict__ ald,
                             int Nn, int F)
{
    const int wave = threadIdx.x >> 5;
    const int lane = threadIdx.x & 31;
    const int n = blockIdx.x * (blockDim.x >> 5) + wave;
    if (n >= Nn) return;
    float s0 = 0.f, s1 = 0.f;
    for (int f = lane; f < F; f += 32) {
        float hv = h[(size_t)n * F + f];
        s0 += hv * asrc[f];
        s1 += hv * adst[f];
    }
    #pragma unroll
    for (int m = 16; m >= 1; m >>= 1) {
        s0 += __shfl_xor(s0, m, 32);
        s1 += __shfl_xor(s1, m, 32);
    }
    if (lane == 0) { als[n] = s0; ald[n] = s1; }
}

// -------- edge pass 1: e = leaky_relu(as[src]+ad[dst]); segment-max via uint key -------
__global__ void edge_pass1(const int* __restrict__ ei,
                           const float* __restrict__ als, const float* __restrict__ ald,
                           float* __restrict__ ew, unsigned* __restrict__ mkey)
{
    int i = blockIdx.x * blockDim.x + threadIdx.x;
    if (i >= E_TOT) return;
    int s, d;
    if (i < N_EDGES) { s = ei[i]; d = ei[N_EDGES + i]; } else { s = d = i - N_EDGES; }
    float e = als[s] + ald[d];
    e = (e > 0.f) ? e : SLOPE_ * e;
    ew[i] = e;
    atomicMax(&mkey[d], fkey(e));
}

// -------- edge pass 2: w = exp(e - max[dst]); segment-sum ------------------------------
__global__ void edge_pass2(const int* __restrict__ ei,
                           float* __restrict__ ew,
                           const unsigned* __restrict__ mkey, float* __restrict__ z)
{
    int i = blockIdx.x * blockDim.x + threadIdx.x;
    if (i >= E_TOT) return;
    int d = (i < N_EDGES) ? ei[N_EDGES + i] : (i - N_EDGES);
    float w = __expf(ew[i] - funkey(mkey[d]));
    ew[i] = w;
    atomicAdd(&z[d], w);
}

// -------- edge pass 3: agg[dst] += (w/z[dst]) * h[src]   (one wave per edge) -----------
__global__ void edge_pass3(const int* __restrict__ ei,
                           const float* __restrict__ ew, const float* __restrict__ z,
                           const float* __restrict__ h, float* __restrict__ agg, int F)
{
    const int wave = threadIdx.x >> 5;
    const int lane = threadIdx.x & 31;
    const int i = blockIdx.x * (blockDim.x >> 5) + wave;
    if (i >= E_TOT) return;
    int s, d;
    if (i < N_EDGES) { s = ei[i]; d = ei[N_EDGES + i]; } else { s = d = i - N_EDGES; }
    if (lane == 0) __builtin_prefetch(h + (size_t)s * F, 0, 0);  // global_prefetch_b8
    const float attn = ew[i] / z[d];
    for (int f = lane; f < F; f += 32)
        atomicAdd(&agg[(size_t)d * F + f], attn * h[(size_t)s * F + f]);
}

// -------- o(bf16) = relu(v + bias): fused activation + conversion ----------------------
__global__ void bias_relu_cvt(const float* __restrict__ v, const float* __restrict__ b,
                              __bf16* __restrict__ o, int total, int F)
{
    int i = blockIdx.x * blockDim.x + threadIdx.x;
    if (i >= total) return;
    float r = v[i] + b[i % F];
    o[i] = f2bf(fmaxf(r, 0.f));
}

// -------- hm[i] = agg2b[mask[i]]  (bf16 gather) ----------------------------------------
__global__ void gather_mask_bf16(const __bf16* __restrict__ src, const int* __restrict__ mask,
                                 __bf16* __restrict__ dst, int total, int F)
{
    int i = blockIdx.x * blockDim.x + threadIdx.x;
    if (i >= total) return;
    int row = i / F, f = i - row * F;
    dst[i] = src[(size_t)mask[row] * F + f];
}

// -------- tiny head: out[MxC] = zfc[Mx128] @ fc2_w[128xC] + fc2_b ----------------------
__global__ void fc2_kernel(const float* __restrict__ zfc, const float* __restrict__ w,
                           const float* __restrict__ b, float* __restrict__ out, int total)
{
    int i = blockIdx.x * blockDim.x + threadIdx.x;
    if (i >= total) return;
    int row = i / C_, c = i - row * C_;
    float acc = b[c];
    #pragma unroll 4
    for (int k = 0; k < FC1_; ++k)
        acc += zfc[(size_t)row * FC1_ + k] * w[k * C_ + c];
    out[i] = acc;
}

static inline int cdiv(long long a, long long b) { return (int)((a + b - 1) / b); }

extern "C" void kernel_launch(void* const* d_in, const int* in_sizes, int n_in,
                              void* d_out, int out_size, void* d_ws, size_t ws_size,
                              hipStream_t stream)
{
    (void)in_sizes; (void)n_in; (void)out_size; (void)ws_size;
    const float* x    = (const float*)d_in[0];
    const int*   ei   = (const int*)  d_in[1];
    const int*   mask = (const int*)  d_in[2];
    const float* W1   = (const float*)d_in[3];
    const float* as1  = (const float*)d_in[4];
    const float* ad1  = (const float*)d_in[5];
    const float* b1   = (const float*)d_in[6];
    const float* W2   = (const float*)d_in[7];
    const float* as2  = (const float*)d_in[8];
    const float* ad2  = (const float*)d_in[9];
    const float* b2   = (const float*)d_in[10];
    const float* fc1w = (const float*)d_in[11];
    const float* fc1b = (const float*)d_in[12];
    const float* fc2w = (const float*)d_in[13];
    const float* fc2b = (const float*)d_in[14];
    float* out = (float*)d_out;

    char* ws = (char*)d_ws;
    size_t off = 0;
    auto alloc = [&](size_t bytes) -> void* {
        void* p = ws + off; off += (bytes + 255) & ~(size_t)255; return p;
    };
    float*    h1    = (float*)   alloc((size_t)N_NODES * H1_ * 4);
    float*    agg1  = (float*)   alloc((size_t)N_NODES * H1_ * 4);
    float*    h2    = (float*)   alloc((size_t)N_NODES * H2_ * 4);
    float*    agg2  = (float*)   alloc((size_t)N_NODES * H2_ * 4);
    __bf16*   xb    = (__bf16*)  alloc((size_t)N_NODES * F_IN * 2);
    __bf16*   agg1b = (__bf16*)  alloc((size_t)N_NODES * H1_ * 2);
    __bf16*   agg2b = (__bf16*)  alloc((size_t)N_NODES * H2_ * 2);
    __bf16*   hmb   = (__bf16*)  alloc((size_t)M_ * H2_ * 2);
    __bf16*   W1T   = (__bf16*)  alloc((size_t)F_IN * H1_ * 2);
    __bf16*   W2T   = (__bf16*)  alloc((size_t)H1_ * H2_ * 2);
    __bf16*   fc1wT = (__bf16*)  alloc((size_t)H2_ * FC1_ * 2);
    float*    als   = (float*)   alloc((size_t)N_NODES * 4);
    float*    ald   = (float*)   alloc((size_t)N_NODES * 4);
    unsigned* mk    = (unsigned*)alloc((size_t)N_NODES * 4);
    float*    z     = (float*)   alloc((size_t)N_NODES * 4);
    float*    ew    = (float*)   alloc((size_t)E_TOT * 4);
    float*    zfc   = (float*)   alloc((size_t)M_ * FC1_ * 4);

    const int TB = 256, WPB = TB / 32;

    // ===== operand conversion (once per call) =====
    cvt_bf16<<<cdiv((long long)N_NODES * F_IN / 4, TB), TB, 0, stream>>>(x, xb, (long long)N_NODES * F_IN);
    cvt_bf16_T<<<cdiv(F_IN * H1_, TB), TB, 0, stream>>>(W1, W1T, F_IN, H1_);
    cvt_bf16_T<<<cdiv(H1_ * H2_, TB), TB, 0, stream>>>(W2, W2T, H1_, H2_);
    cvt_bf16_T<<<cdiv(H2_ * FC1_, TB), TB, 0, stream>>>(fc1w, fc1wT, H2_, FC1_);

    // ===== Layer 1 =====
    gemm_bf16_wmma<128, 8><<<cdiv(N_NODES / 16, WPB), TB, 0, stream>>>(xb, W1T, h1, nullptr, N_NODES, 0);
    alpha_kernel<<<cdiv(N_NODES, WPB), TB, 0, stream>>>(h1, as1, ad1, als, ald, N_NODES, H1_);
    hipMemsetAsync(mk,   0, (size_t)N_NODES * 4, stream);
    hipMemsetAsync(z,    0, (size_t)N_NODES * 4, stream);
    hipMemsetAsync(agg1, 0, (size_t)N_NODES * H1_ * 4, stream);
    edge_pass1<<<cdiv(E_TOT, TB), TB, 0, stream>>>(ei, als, ald, ew, mk);
    edge_pass2<<<cdiv(E_TOT, TB), TB, 0, stream>>>(ei, ew, mk, z);
    edge_pass3<<<cdiv(E_TOT, WPB), TB, 0, stream>>>(ei, ew, z, h1, agg1, H1_);
    bias_relu_cvt<<<cdiv((long long)N_NODES * H1_, TB), TB, 0, stream>>>(agg1, b1, agg1b, N_NODES * H1_, H1_);

    // ===== Layer 2 =====
    gemm_bf16_wmma<128, 4><<<cdiv(N_NODES / 16, WPB), TB, 0, stream>>>(agg1b, W2T, h2, nullptr, N_NODES, 0);
    alpha_kernel<<<cdiv(N_NODES, WPB), TB, 0, stream>>>(h2, as2, ad2, als, ald, N_NODES, H2_);
    hipMemsetAsync(mk,   0, (size_t)N_NODES * 4, stream);
    hipMemsetAsync(z,    0, (size_t)N_NODES * 4, stream);
    hipMemsetAsync(agg2, 0, (size_t)N_NODES * H2_ * 4, stream);
    edge_pass1<<<cdiv(E_TOT, TB), TB, 0, stream>>>(ei, als, ald, ew, mk);
    edge_pass2<<<cdiv(E_TOT, TB), TB, 0, stream>>>(ei, ew, mk, z);
    edge_pass3<<<cdiv(E_TOT, WPB), TB, 0, stream>>>(ei, ew, z, h2, agg2, H2_);
    bias_relu_cvt<<<cdiv((long long)N_NODES * H2_, TB), TB, 0, stream>>>(agg2, b2, agg2b, N_NODES * H2_, H2_);

    // ===== FC head on masked nodes =====
    gather_mask_bf16<<<cdiv((long long)M_ * H2_, TB), TB, 0, stream>>>(agg2b, mask, hmb, M_ * H2_, H2_);
    gemm_bf16_wmma<64, 8><<<cdiv(M_ / 16, WPB), TB, 0, stream>>>(hmb, fc1wT, zfc, fc1b, M_, 1);
    fc2_kernel<<<cdiv((long long)M_ * C_, TB), TB, 0, stream>>>(zfc, fc2w, fc2b, out, M_ * C_);
}
